// NAIS_basic_65240553226371
// MI455X (gfx1250) — matrix operand order repaired and checked by reading, not compiled, not run on hardware
//
#include <hip/hip_runtime.h>
#include <hip/hip_bf16.h>
#include <math.h>

typedef _Float16 f16;
typedef _Float16 v4h  __attribute__((ext_vector_type(4)));
typedef _Float16 v8h  __attribute__((ext_vector_type(8)));
typedef _Float16 v16h __attribute__((ext_vector_type(16)));
typedef float    v8f  __attribute__((ext_vector_type(8)));

#define ITEM_NUM 100000
#define EMBED    128
#define HIDDEN   256
#define B_SZ     4096
#define H_SZ     200
#define MT       13          // ceil(H/16)
#define MPAD     (MT * 16)   // 208
#define KSTEPS   (EMBED / 32)
#define NTILES   (HIDDEN / 16)
#define XSTR     136         // padded LDS row stride (halves) -> 272B
#define WSTR     136

// --- one-time prep: W1 [128,256] f32 -> W1^T [256,128] f16 in workspace ---
__global__ void nais_prep_w1t(const float* __restrict__ W1, f16* __restrict__ W1T) {
    int idx = blockIdx.x * blockDim.x + threadIdx.x;
    if (idx < EMBED * HIDDEN) {
        int n = idx >> 7;      // row of W1T (hidden unit)
        int k = idx & 127;     // col of W1T (embed dim)
        W1T[idx] = (f16)W1[k * HIDDEN + n];
    }
}

// --- main fused kernel: one workgroup per batch row ---
__launch_bounds__(256)
__global__ void nais_main(const int*   __restrict__ history,
                          const int*   __restrict__ target,
                          const float* __restrict__ embH,
                          const float* __restrict__ embT,
                          const f16*   __restrict__ W1T,
                          const float* __restrict__ b1,
                          const float* __restrict__ w2,
                          float*       __restrict__ out)
{
    __shared__ f16   xs[MPAD * XSTR];      // 56,576 B : X = hist*tgt (fp16, A-matrix)
    __shared__ f16   ws[HIDDEN * WSTR];    // 69,632 B : W1^T (fp16, B-matrix source)
    __shared__ float tgt[EMBED];
    __shared__ float b1s[HIDDEN];
    __shared__ float w2s[HIDDEN];
    __shared__ float abuf[MPAD];           // a[b,h]
    __shared__ float dotb[MPAD];           // hist_e . tgt_e
    __shared__ float red[16];

    const int b      = blockIdx.x;
    const int tid    = threadIdx.x;
    const int lane   = tid & 31;
    const int wave   = tid >> 5;
    const int halfId = lane >> 4;          // 0: lanes 0-15, 1: lanes 16-31
    const int lq     = lane & 15;
    const int tgt_item = target[b];

    // target embedding + b1/w2 -> LDS
    if (tid < EMBED) tgt[tid] = embT[(long)tgt_item * EMBED + tid];
    b1s[tid] = b1[tid];
    w2s[tid] = w2[tid];

    // W1^T -> LDS (each thread copies one 256-byte row as 16x int4)
    {
        const int4* src = (const int4*)(W1T + (long)tid * EMBED);
        f16* dst = ws + tid * WSTR;
        #pragma unroll
        for (int c = 0; c < 16; ++c) *(int4*)&dst[c * 8] = src[c];
    }
    __syncthreads();

    // -------- gather + build X (fp16) + dot[m] --------
    // 8 threads per row, each thread: 16 consecutive k (4 x float4)
    const int sub   = tid & 7;
    const int rbase = tid >> 3;
    for (int m = rbase; m < MPAD; m += 32) {
        float partial = 0.f;
        f16* xrow = xs + m * XSTR;
        if (m < H_SZ) {
            int item = history[(long)b * H_SZ + m];
            const float4* hrow = (const float4*)(embH + (long)item * EMBED);
            const float4* trow = (const float4*)tgt;
            #pragma unroll
            for (int c = 0; c < 4; ++c) {
                int q = sub * 4 + c;
                float4 hv = hrow[q];
                float4 tv = trow[q];
                float x0 = hv.x * tv.x, x1 = hv.y * tv.y;
                float x2 = hv.z * tv.z, x3 = hv.w * tv.w;
                partial += (x0 + x1) + (x2 + x3);
                v4h pk = { (f16)x0, (f16)x1, (f16)x2, (f16)x3 };
                *(v4h*)&xrow[q * 4] = pk;
            }
        } else {
            v4h z = {};
            #pragma unroll
            for (int c = 0; c < 4; ++c) *(v4h*)&xrow[(sub * 4 + c) * 4] = z;
        }
        partial += __shfl_xor(partial, 1);
        partial += __shfl_xor(partial, 2);
        partial += __shfl_xor(partial, 4);
        if (sub == 0) dotb[m] = partial;
    }
    __syncthreads();

    // -------- WMMA GEMM + fused bias/ReLU/dot-with-w2 epilogue --------
    // Each wave OWNS M-tiles (mt = wave, wave+8) and sweeps all 16 N-tiles,
    // accumulating a[m] in registers -> one shuffle-reduce per M-tile, no atomics.
    #pragma unroll 1
    for (int mt = wave; mt < MT; mt += 8) {
        // A fragments: 16x32 f16 layout -> per lane two contiguous 8-half chunks
        v16h afr[KSTEPS];
        const f16* xrow = xs + (mt * 16 + lq) * XSTR;
        #pragma unroll
        for (int kk = 0; kk < KSTEPS; ++kk) {
            union { v16h v; v8h h[2]; } u;
            u.h[0] = *(const v8h*)&xrow[kk * 32 +      8 * halfId];
            u.h[1] = *(const v8h*)&xrow[kk * 32 + 16 + 8 * halfId];
            afr[kk] = u.v;
        }

        float acc[8];
        #pragma unroll
        for (int r = 0; r < 8; ++r) acc[r] = 0.f;

        // keep this loop ROLLED: full unroll spills B-frags to scratch
        #pragma unroll 1
        for (int ntp = 0; ntp < NTILES; ntp += 2) {   // two independent C chains
            const int n0 = ntp * 16 + lq;
            const int n1 = n0 + 16;
            const float b1v0 = b1s[n0], w2v0 = w2s[n0];
            const float b1v1 = b1s[n1], w2v1 = w2s[n1];
            const f16* wrow0 = ws + n0 * WSTR;
            const f16* wrow1 = ws + n1 * WSTR;
            v8f c0 = {}, c1 = {};
            #pragma unroll
            for (int kk = 0; kk < KSTEPS; ++kk) {
                union { v16h v; v8h h[2]; } ub0, ub1;  // B 32x16: two 8-half k-chunks (same order as A)
                ub0.h[0] = *(const v8h*)&wrow0[kk * 32 +      8 * halfId];
                ub0.h[1] = *(const v8h*)&wrow0[kk * 32 + 16 + 8 * halfId];
                ub1.h[0] = *(const v8h*)&wrow1[kk * 32 +      8 * halfId];
                ub1.h[1] = *(const v8h*)&wrow1[kk * 32 + 16 + 8 * halfId];
                c0 = __builtin_amdgcn_wmma_f32_16x16x32_f16(
                         false, afr[kk], false, ub0.v, (short)0, c0, false, false);
                c1 = __builtin_amdgcn_wmma_f32_16x16x32_f16(
                         false, afr[kk], false, ub1.v, (short)0, c1, false, false);
            }
            #pragma unroll
            for (int r = 0; r < 8; ++r) {
                float v0 = c0[r] + b1v0;  v0 = v0 > 0.f ? v0 : 0.f;
                float v1 = c1[r] + b1v1;  v1 = v1 > 0.f ? v1 : 0.f;
                acc[r] += v0 * w2v0 + v1 * w2v1;
            }
        }
        // reduce acc across the 16 lanes of each half (columns n within lane group)
        #pragma unroll
        for (int r = 0; r < 8; ++r) {
            float v = acc[r];
            v += __shfl_xor(v, 1);
            v += __shfl_xor(v, 2);
            v += __shfl_xor(v, 4);
            v += __shfl_xor(v, 8);
            if (lq == 0) abuf[mt * 16 + r + 8 * halfId] = v;
        }
    }
    __syncthreads();

    // -------- exp / mask / beta-normalize / sigmoid --------
    float e = 0.f, ed = 0.f;
    if (tid < H_SZ) {
        int item = history[(long)b * H_SZ + tid];
        float msk = (item != tgt_item) ? 1.0f : 0.0f;
        float ev  = __expf(abuf[tid]) * msk;
        e  = ev;
        ed = ev * dotb[tid];
    }
    #pragma unroll
    for (int m = 1; m < 32; m <<= 1) {
        e  += __shfl_xor(e,  m);
        ed += __shfl_xor(ed, m);
    }
    if (lane == 0) { red[wave] = e; red[8 + wave] = ed; }
    __syncthreads();
    if (tid == 0) {
        float S1 = 0.f, S2 = 0.f;
        #pragma unroll
        for (int w = 0; w < 8; ++w) { S1 += red[w]; S2 += red[8 + w]; }
        S1 = fmaxf(S1, 1e-30f);
        float pred = S2 * rsqrtf(S1);          // denom = S1^BETA, BETA = 0.5
        out[b] = 1.0f / (1.0f + __expf(-pred));
    }
}

extern "C" void kernel_launch(void* const* d_in, const int* in_sizes, int n_in,
                              void* d_out, int out_size, void* d_ws, size_t ws_size,
                              hipStream_t stream) {
    (void)in_sizes; (void)n_in; (void)out_size; (void)ws_size;
    const int*   history = (const int*)  d_in[0];
    const int*   target  = (const int*)  d_in[1];
    const float* embH    = (const float*)d_in[2];
    const float* embT    = (const float*)d_in[3];
    const float* W1      = (const float*)d_in[4];
    const float* b1      = (const float*)d_in[5];
    const float* w2      = (const float*)d_in[6];
    float* out = (float*)d_out;
    f16*   W1T = (f16*)d_ws;   // 64 KB

    nais_prep_w1t<<<(EMBED * HIDDEN + 255) / 256, 256, 0, stream>>>(W1, W1T);
    nais_main<<<B_SZ, 256, 0, stream>>>(history, target, embH, embT, W1T, b1, w2, out);
}
